// MultiExpertActor_35759897706830
// MI455X (gfx1250) — compile-verified, matrix operand order checked by reading, not run on hardware
//
#include <hip/hip_runtime.h>
#include <hip/hip_bf16.h>

// MI455X / gfx1250: bf16 WMMA (16x16x32, f32 accum) everywhere; all weights
// pre-transposed to [N,K] bf16 so WMMA B-fragments are direct 16B global loads
// (L2-resident); A activation tiles staged into LDS with CDNA5 async-to-LDS
// copies (GLOBAL_LOAD_ASYNC_TO_LDS_B128 + s_wait_asynccnt), double-buffered,
// one workgroup barrier per pipeline stage.

typedef __bf16 bf16;
typedef __attribute__((ext_vector_type(16))) __bf16 v16bf;
typedef __attribute__((ext_vector_type(8)))  __bf16 v8bf;
typedef __attribute__((ext_vector_type(8)))  float  v8f;
typedef __attribute__((ext_vector_type(4)))  float  v4f;

#define BATCH   65536
#define NEXP    16

union V16U { v16bf v; v8bf h[2]; };

static __device__ __forceinline__ v16bf make_frag(v8bf lo, v8bf hi) {
    V16U u; u.h[0] = lo; u.h[1] = hi; return u.v;
}
static __device__ __forceinline__ v8f zero8() {
    v8f z = {0.f,0.f,0.f,0.f,0.f,0.f,0.f,0.f}; return z;
}

// ---- CDNA5 async global->LDS copy (ASYNCcnt-tracked) ----
static __device__ __forceinline__ void async_copy_b128(void* lds_generic, const void* gaddr) {
    unsigned lds_off = (unsigned)(unsigned long long)lds_generic;  // addr[31:0] = LDS byte offset
    asm volatile("global_load_async_to_lds_b128 %0, %1, off"
                 :: "v"(lds_off), "v"(gaddr) : "memory");
}
template<int N> static __device__ __forceinline__ void wait_asynccnt() {
#if __has_builtin(__builtin_amdgcn_s_wait_asynccnt)
    __builtin_amdgcn_s_wait_asynccnt((short)N);
#else
    asm volatile("s_wait_asynccnt %0" :: "i"(N) : "memory");
#endif
}

// ---------------- conversion kernels ----------------

__global__ __launch_bounds__(256) void cvt_f32_to_bf16(const float* __restrict__ s,
                                                       bf16* __restrict__ d, long n) {
    long i = ((long)blockIdx.x * 256 + threadIdx.x) * 4;
    if (i + 3 < n) {
        v4f v = *(const v4f*)(s + i);
        d[i + 0] = (bf16)v.x; d[i + 1] = (bf16)v.y;
        d[i + 2] = (bf16)v.z; d[i + 3] = (bf16)v.w;
    } else {
        for (long j = i; j < n; ++j) d[j] = (bf16)s[j];
    }
}

// src [E][K][N] f32  ->  dst [E][N][K] bf16   (weights are tiny; one-time)
__global__ __launch_bounds__(256) void transpose_cvt(const float* __restrict__ src,
                                                     bf16* __restrict__ dst,
                                                     int E, int K, int N) {
    long i = (long)blockIdx.x * 256 + threadIdx.x;
    long total = (long)E * K * N;
    if (i >= total) return;
    int e = (int)(i / ((long)K * N));
    long r = i - (long)e * K * N;
    int k = (int)(r / N);
    int n = (int)(r - (long)k * N);
    dst[((long)e * N + n) * K + k] = (bf16)src[i];
}

// ---------------- tiled WMMA GEMM: C = act(A*BT^T + bias) ----------------
// A [M,K] bf16 row-major (streamed, staged async into LDS, double-buffered),
// BT [N,K] bf16 (L2-resident, fragments read straight from global),
// bias [N] f32. Block tile 64x128, 8 waves (2x4), each wave 32x32, K-step 32.
// Single barrier per K-step: wait(own copies) -> barrier -> issue(next) -> compute.
template<bool RELU, bool OUTBF>
__global__ __launch_bounds__(256) void gemm_bias_act(const bf16* __restrict__ A,
                                                     const bf16* __restrict__ BT,
                                                     const float* __restrict__ bias,
                                                     void* __restrict__ C,
                                                     int M, int N, int K) {
    __shared__ __align__(16) bf16 As[2][64][40];   // 64x32 (+8 pad), 80B row stride
    const int t    = threadIdx.x;
    const int m0   = blockIdx.y * 64;
    const int n0   = blockIdx.x * 128;
    const int wv   = t >> 5, lane = t & 31;
    const int wm   = wv & 1,  wn  = wv >> 1;       // 2 x 4 wave grid
    const int kh   = lane >> 4, lr = lane & 15;

    const int arow = t >> 2,  akc = (t & 3) * 8;   // 256 thr x 8 elems = 64x32 tile
    const int T = K >> 5;

    // prologue: async-stage A tile 0
    async_copy_b128(&As[0][arow][akc], &A[(long)(m0 + arow) * K + akc]);

    v8f acc[2][2] = { { zero8(), zero8() }, { zero8(), zero8() } };

    for (int ks = 0; ks < T; ++ks) {
        wait_asynccnt<0>();      // tile ks landed (issued one compute-phase ago)
        __syncthreads();         // all waves' copies done + prev reads of next buf done
        if (ks + 1 < T)          // overlap tile ks+1 copy with tile ks compute
            async_copy_b128(&As[(ks + 1) & 1][arow][akc],
                            &A[(long)(m0 + arow) * K + (ks + 1) * 32 + akc]);

        const int buf = ks & 1;
        v16bf af[2], bfrag[2];
        #pragma unroll
        for (int i = 0; i < 2; ++i) {
            int r = wm * 32 + i * 16 + lr;
            af[i] = make_frag(*(const v8bf*)&As[buf][r][kh * 8],
                              *(const v8bf*)&As[buf][r][16 + kh * 8]);
        }
        #pragma unroll
        for (int j = 0; j < 2; ++j) {
            int n = n0 + wn * 32 + j * 16 + lr;
            const bf16* p = BT + (long)n * K + ks * 32 + kh * 8;
            bfrag[j] = make_frag(*(const v8bf*)p, *(const v8bf*)(p + 16));
        }
        #pragma unroll
        for (int i = 0; i < 2; ++i)
            #pragma unroll
            for (int j = 0; j < 2; ++j)
                acc[i][j] = __builtin_amdgcn_wmma_f32_16x16x32_bf16(
                    false, af[i], false, bfrag[j], (short)0, acc[i][j], false, false);
    }

    #pragma unroll
    for (int i = 0; i < 2; ++i) {
        #pragma unroll
        for (int j = 0; j < 2; ++j) {
            int col = n0 + wn * 32 + j * 16 + lr;
            float bv = bias[col];
            #pragma unroll
            for (int r = 0; r < 8; ++r) {
                int row = m0 + wm * 32 + i * 16 + kh * 8 + r;
                float v = acc[i][j][r] + bv;
                if (RELU) v = fmaxf(v, 0.0f);
                if (OUTBF) ((bf16*)C)[(long)row * N + col] = (bf16)v;
                else       ((float*)C)[(long)row * N + col] = v;
            }
        }
    }
}

// ---------------- selector head: s1[B,128] @ Ws2[128,16] + softmax ----------------
__global__ __launch_bounds__(256) void selector_head(const bf16* __restrict__ S1,
                                                     const float* __restrict__ Ws2,
                                                     const float* __restrict__ bs2,
                                                     float* __restrict__ Wout) {
    __shared__ __align__(16) bf16 S1s[256][136];  // 272B row stride (16B aligned)
    __shared__ float W2s[128 * 16];
    const int t = threadIdx.x;
    const long b0 = (long)blockIdx.x * 256;
    for (int i = t; i < 128 * 16; i += 256) W2s[i] = Ws2[i];
    for (int c = t; c < 256 * 16; c += 256) {        // coalesced 8-elem chunks
        int row = c >> 4, col = (c & 15) * 8;
        async_copy_b128(&S1s[row][col], &S1[(b0 + row) * 128 + col]);
    }
    wait_asynccnt<0>();
    __syncthreads();

    float s[16];
    #pragma unroll
    for (int j = 0; j < 16; ++j) s[j] = bs2[j];
    for (int k = 0; k < 128; ++k) {
        float x = (float)S1s[t][k];
        #pragma unroll
        for (int j = 0; j < 16; ++j) s[j] = fmaf(x, W2s[k * 16 + j], s[j]);
    }
    float mx = s[0];
    #pragma unroll
    for (int j = 1; j < 16; ++j) mx = fmaxf(mx, s[j]);
    float sum = 0.f;
    #pragma unroll
    for (int j = 0; j < 16; ++j) { s[j] = __expf(s[j] - mx); sum += s[j]; }
    float inv = 1.0f / sum;
    #pragma unroll
    for (int j = 0; j < 16; ++j) Wout[(b0 + t) * 16 + j] = s[j] * inv;
}

// ---------------- fused experts + softmax-weighted fusion ----------------
// H [B,512] bf16; We1T [16][256][512]; We2T [16][32][256]; be1/be2 f32;
// Wsel [B,16] f32; Out [B,32] f32. 64-row h-tile resident in LDS for all
// experts; EH staged through a double-buffered LDS tile (1 barrier/expert).
__global__ __launch_bounds__(256) void experts_fused(const bf16* __restrict__ H,
                                                     const bf16* __restrict__ We1T,
                                                     const float* __restrict__ be1,
                                                     const bf16* __restrict__ We2T,
                                                     const float* __restrict__ be2,
                                                     const float* __restrict__ Wsel,
                                                     float* __restrict__ Out) {
    __shared__ __align__(16) bf16 Hs[64][520];      // 64x512 (+8 pad) = 66.5 KB
    __shared__ __align__(16) bf16 EHs[2][64][264];  // 2x (64x256 +8 pad) = 67.6 KB
    __shared__ float Wls[64][16];
    const int t = threadIdx.x;
    const long b0 = (long)blockIdx.x * 64;

    { // async-stage the whole h tile (read 16x per expert loop afterwards)
        int row = t >> 2, cb = (t & 3) * 128;
        #pragma unroll
        for (int i = 0; i < 16; ++i)
            async_copy_b128(&Hs[row][cb + i * 8], &H[(b0 + row) * 512 + cb + i * 8]);
    }
    for (int i = t; i < 64 * 16; i += 256)
        Wls[i >> 4][i & 15] = Wsel[(b0 + (i >> 4)) * 16 + (i & 15)];
    wait_asynccnt<0>();
    __syncthreads();

    const int wv = t >> 5, lane = t & 31;
    const int sub = wv >> 1, ch = wv & 1;        // 4 row-subtiles x 2 col-halves
    const int kh = lane >> 4, lr = lane & 15;
    const int r0 = sub * 16;

    v8f facc = zero8();

    for (int e = 0; e < NEXP; ++e) {
        const int ebuf = e & 1;
        // ---- GEMM1: eh[r0..r0+15, ch*128..+127] = relu(h @ We1[e] + be1[e]) ----
        v8f acc1[8] = { zero8(), zero8(), zero8(), zero8(),
                        zero8(), zero8(), zero8(), zero8() };
        const bf16* W1e = We1T + (long)e * 256 * 512;
        for (int k0 = 0; k0 < 512; k0 += 32) {
            int ar = r0 + lr;
            v16bf af = make_frag(*(const v8bf*)&Hs[ar][k0 + kh * 8],
                                 *(const v8bf*)&Hs[ar][k0 + 16 + kh * 8]);
            #pragma unroll
            for (int jt = 0; jt < 8; ++jt) {
                int n = ch * 128 + jt * 16 + lr;
                const bf16* p = W1e + (long)n * 512 + k0 + kh * 8;
                v16bf bfrag = make_frag(*(const v8bf*)p, *(const v8bf*)(p + 16));
                acc1[jt] = __builtin_amdgcn_wmma_f32_16x16x32_bf16(
                    false, af, false, bfrag, (short)0, acc1[jt], false, false);
            }
        }
        #pragma unroll
        for (int jt = 0; jt < 8; ++jt) {
            int col = ch * 128 + jt * 16 + lr;
            float bv = be1[e * 256 + col];
            #pragma unroll
            for (int r = 0; r < 8; ++r) {
                int row = r0 + kh * 8 + r;
                EHs[ebuf][row][col] = (bf16)fmaxf(acc1[jt][r] + bv, 0.0f);
            }
        }
        __syncthreads();   // EH tile e complete; also fences e-1's readers of other buf

        // ---- GEMM2: ea[r0..r0+15, ch*16..+15] = eh @ We2[e] + be2[e] ----
        v8f acc2 = zero8();
        #pragma unroll
        for (int k0 = 0; k0 < 256; k0 += 32) {
            int ar = r0 + lr;
            v16bf af = make_frag(*(const v8bf*)&EHs[ebuf][ar][k0 + kh * 8],
                                 *(const v8bf*)&EHs[ebuf][ar][k0 + 16 + kh * 8]);
            int a = ch * 16 + lr;
            const bf16* p = We2T + ((long)e * 32 + a) * 256 + k0 + kh * 8;
            v16bf bfrag = make_frag(*(const v8bf*)p, *(const v8bf*)(p + 16));
            acc2 = __builtin_amdgcn_wmma_f32_16x16x32_bf16(
                false, af, false, bfrag, (short)0, acc2, false, false);
        }
        int col = ch * 16 + lr;
        float b2 = be2[e * 32 + col];
        #pragma unroll
        for (int r = 0; r < 8; ++r) {
            int row = r0 + kh * 8 + r;
            facc[r] += Wls[row][e] * (acc2[r] + b2);
        }
        // no trailing barrier: next expert writes the other EH buffer
    }

    int col = ch * 16 + lr;
    #pragma unroll
    for (int r = 0; r < 8; ++r) {
        long row = b0 + r0 + kh * 8 + r;
        Out[row * 32 + col] = facc[r];
    }
}

// ---------------- host-side launch ----------------

extern "C" void kernel_launch(void* const* d_in, const int* in_sizes, int n_in,
                              void* d_out, int out_size, void* d_ws, size_t ws_size,
                              hipStream_t stream) {
    const float* obs = (const float*)d_in[0];
    const float* Wb0 = (const float*)d_in[1];
    const float* bb0 = (const float*)d_in[2];
    const float* Wb1 = (const float*)d_in[3];
    const float* bb1 = (const float*)d_in[4];
    const float* Wb2 = (const float*)d_in[5];
    const float* bb2 = (const float*)d_in[6];
    const float* We1 = (const float*)d_in[7];
    const float* be1 = (const float*)d_in[8];
    const float* We2 = (const float*)d_in[9];
    const float* be2 = (const float*)d_in[10];
    const float* Ws0 = (const float*)d_in[11];
    const float* bs0 = (const float*)d_in[12];
    const float* Ws1 = (const float*)d_in[13];
    const float* bs1 = (const float*)d_in[14];
    const float* Ws2 = (const float*)d_in[15];
    const float* bs2 = (const float*)d_in[16];
    float* out = (float*)d_out;

    char* ws = (char*)d_ws;
    size_t off = 0;
    auto take = [&](size_t bytes) -> char* {
        char* p = ws + off;
        off = (off + bytes + 255) & ~(size_t)255;
        return p;
    };

    const long B = BATCH;
    bf16* obs_b = (bf16*)take(B * 512 * 2);
    bf16* wb0t  = (bf16*)take((long)1024 * 512 * 2);   // [N=1024][K=512]
    bf16* wb1t  = (bf16*)take((long)1024 * 1024 * 2);  // [1024][1024]
    bf16* wb2t  = (bf16*)take((long)512 * 1024 * 2);   // [512][1024]
    bf16* ws0t  = (bf16*)take((long)256 * 512 * 2);    // [256][512]
    bf16* ws1t  = (bf16*)take((long)128 * 256 * 2);    // [128][256]
    bf16* we1t  = (bf16*)take((long)16 * 256 * 512 * 2);
    bf16* we2t  = (bf16*)take((long)16 * 32 * 256 * 2);
    bf16* h0    = (bf16*)take(B * 1024 * 2);
    bf16* h1    = (bf16*)take(B * 1024 * 2);
    bf16* hb    = (bf16*)take(B * 512 * 2);
    bf16* s0    = (bf16*)take(B * 256 * 2);
    bf16* s1    = (bf16*)take(B * 128 * 2);
    float* wsel = (float*)take(B * 16 * 4);
    (void)ws_size; (void)n_in; (void)in_sizes; (void)out_size;

    // conversions / weight transposes (one-time, tiny)
    {
        long n = B * 512;
        cvt_f32_to_bf16<<<(int)((n + 1023) / 1024), 256, 0, stream>>>(obs, obs_b, n);
    }
    auto tcvt = [&](const float* s, bf16* d, int E, int K, int N) {
        long tot = (long)E * K * N;
        transpose_cvt<<<(int)((tot + 255) / 256), 256, 0, stream>>>(s, d, E, K, N);
    };
    tcvt(Wb0, wb0t, 1, 512, 1024);
    tcvt(Wb1, wb1t, 1, 1024, 1024);
    tcvt(Wb2, wb2t, 1, 1024, 512);
    tcvt(Ws0, ws0t, 1, 512, 256);
    tcvt(Ws1, ws1t, 1, 256, 128);
    tcvt(We1, we1t, 16, 512, 256);
    tcvt(We2, we2t, 16, 256, 32);

    // backbone
    gemm_bias_act<true, true><<<dim3(1024 / 128, BATCH / 64), 256, 0, stream>>>(
        obs_b, wb0t, bb0, (void*)h0, BATCH, 1024, 512);
    gemm_bias_act<true, true><<<dim3(1024 / 128, BATCH / 64), 256, 0, stream>>>(
        h0, wb1t, bb1, (void*)h1, BATCH, 1024, 1024);
    gemm_bias_act<true, true><<<dim3(512 / 128, BATCH / 64), 256, 0, stream>>>(
        h1, wb2t, bb2, (void*)hb, BATCH, 512, 1024);

    // selector trunk
    gemm_bias_act<true, true><<<dim3(256 / 128, BATCH / 64), 256, 0, stream>>>(
        obs_b, ws0t, bs0, (void*)s0, BATCH, 256, 512);
    gemm_bias_act<true, true><<<dim3(128 / 128, BATCH / 64), 256, 0, stream>>>(
        s0, ws1t, bs1, (void*)s1, BATCH, 128, 256);

    // selector head + softmax
    selector_head<<<BATCH / 256, 256, 0, stream>>>(s1, Ws2, bs2, wsel);

    // fused experts + weighted fusion
    experts_fused<<<BATCH / 64, 256, 0, stream>>>(hb, we1t, be1, we2t, be2, wsel, out);
}
